// BioAttentionFusion_51427938402925
// MI455X (gfx1250) — matrix operand order repaired and checked by reading, not compiled, not run on hardware
//
#include <hip/hip_runtime.h>
#include <math.h>

// ---------------------------------------------------------------------------
// Types for CDNA5 WMMA (gfx1250, wave32): A/B = 16 bf16 per lane (8 VGPRs),
// C/D = 8 f32 per lane.
// ---------------------------------------------------------------------------
typedef __attribute__((ext_vector_type(16))) __bf16          v16bf;
typedef __attribute__((ext_vector_type(8)))  float           v8f;
typedef __attribute__((ext_vector_type(4)))  float           f4;
typedef __attribute__((ext_vector_type(8)))  unsigned short  u16x8;
typedef __attribute__((ext_vector_type(4)))  unsigned int    u32x4;

#define WMMA_BF16(a, b, c) \
    __builtin_amdgcn_wmma_f32_16x16x32_bf16(false, (a), false, (b), (short)0, (c), false, false)

union Frag16 {
    v16bf v;
    unsigned short h[16];
    u16x8 u8[2];
};

// native convert -> v_cvt bf16 (RNE)
__device__ __forceinline__ unsigned short f2bf(float f) {
    __bf16 h = (__bf16)f;
    return __builtin_bit_cast(unsigned short, h);
}

__device__ __forceinline__ v8f zero8() {
    v8f z;
#pragma unroll
    for (int i = 0; i < 8; ++i) z[i] = 0.f;
    return z;
}

// select a whole 8x bf16 vector with 32-bit cndmasks
__device__ __forceinline__ u16x8 sel8(bool valid, u16x8 t) {
    u32x4 w = __builtin_bit_cast(u32x4, t);
#pragma unroll
    for (int j = 0; j < 4; ++j) w[j] = valid ? w[j] : 0u;
    return __builtin_bit_cast(u16x8, w);
}

// ---------------------------------------------------------------------------
// Fragment layouts (CDNA5 ISA §7.12.2), wave32:
//  A 16x32 (MxK): lane = m + 16*half ; K(idx) = (idx&7) + 16*(idx>>3) + 8*half
//  B 32x16 (KxN): lane = n + 16*half ; this code loads K(idx) = idx + 16*half (A too)
//  C 16x16 f32 : VGPR r -> row r + 8*half, col = lane&15
// ---------------------------------------------------------------------------

// A frag from bf16 row-major [*, ld], two 16B runs per lane.
__device__ __forceinline__ v16bf loadA_bf16(const unsigned short* A, int ld, int m0, int lane) {
    int m = m0 + (lane & 15);
    int half = lane >> 4;
    const unsigned short* base = A + (long)m * ld;
    Frag16 fr;
    fr.u8[0] = *(const u16x8*)(base + (half << 4));        // k = idx + 16*half
    fr.u8[1] = *(const u16x8*)(base + 8 + (half << 4));    // k = idx + 16*half
    return fr.v;
}

// B frag from bf16 weights (N x K row-major): 16 contiguous bf16 per lane.
__device__ __forceinline__ v16bf loadB_Wh(const unsigned short* W, int ldw, int n0, int k0,
                                          int lane) {
    int n = n0 + (lane & 15);
    int half = lane >> 4;
    const unsigned short* base = W + (long)n * ldw + k0 + (half << 4);
    Frag16 fr;
    fr.u8[0] = *(const u16x8*)(base);
    fr.u8[1] = *(const u16x8*)(base + 8);
    return fr.v;
}

// B frag of K^T for attention from bf16 K [S,32]; MASKED adds clamp + select.
template <bool MASKED>
__device__ __forceinline__ v16bf loadB_keys(const unsigned short* Kmat, int key0, int S,
                                            int lane) {
    int n = key0 + (lane & 15);
    int half = lane >> 4;
    int nc = MASKED ? (n < (S - 1) ? n : (S - 1)) : n;
    const unsigned short* base = Kmat + (long)nc * 32 + (half << 4);
    u16x8 a = *(const u16x8*)(base);
    u16x8 b = *(const u16x8*)(base + 8);
    Frag16 fr;
    if (MASKED) {
        bool valid = n < S;
        fr.u8[0] = sel8(valid, a);
        fr.u8[1] = sel8(valid, b);
    } else {
        fr.u8[0] = a;
        fr.u8[1] = b;
    }
    return fr.v;
}

// A frag from LDS bf16 tile, row-major 16x32 (ushort), row stride 32.
__device__ __forceinline__ v16bf loadA_lds(const unsigned short* lds, int lane) {
    int m = lane & 15;
    int half = lane >> 4;
    Frag16 fr;
#pragma unroll
    for (int idx = 0; idx < 16; ++idx) {
        int kk = idx + (half << 4);
        fr.h[idx] = lds[m * 32 + kk];
    }
    return fr.v;
}

// B frag from LDS bf16 tile (32 x ld, padded), B[k][n] = lds[k*ld + n0+n].
__device__ __forceinline__ v16bf loadB_lds(const unsigned short* lds, int ld, int n0, int lane) {
    int n = n0 + (lane & 15);
    int half = lane >> 4;
    Frag16 fr;
#pragma unroll
    for (int idx = 0; idx < 16; ++idx) {
        int k = idx + (half << 4);
        fr.h[idx] = lds[k * ld + n];
    }
    return fr.v;
}

__device__ __forceinline__ float gelu_exact(float x) {
    return 0.5f * x * (1.0f + erff(x * 0.70710678118654752f));
}

// ---------------------------------------------------------------------------
// Generic WMMA GEMM: Y[m,n] = sum_k A[m,k] * W[n,k]  (+bias, +epilogue)
// A is f32 (activations), staged per k-step through LDS as bf16 (shared by
// all 8 waves). W is pre-converted bf16, read as b128 runs.
// AMODE 0: A row-major (m,k) -> A[m*lda+k]
// AMODE 1: A k-major   (m,k) -> A[k*lda+m]  (conv1x1 over (C,S) tensors)
// EP 0: +bias -> Y0 row-major (M x N)      EP 1: +bias, GELU -> row-major
// EP 2: +bias -> Y0 col-major Y0[n*M+m]    EP 3: QKV scatter -> Y0/Y1/Y2
// Split-K over two A tensors at k = ksplit (concat in f1).
// ---------------------------------------------------------------------------
template <int AMODE, int EP>
__global__ __launch_bounds__(256) void gemm_bf16_wmma(
    const float* __restrict__ A, const float* __restrict__ A2, int ksplit,
    int lda, int lda2, long strideA, long strideA2,
    const unsigned short* __restrict__ W, const float* __restrict__ bias,
    float* __restrict__ Y0, float* __restrict__ Y1, float* __restrict__ Y2,
    long strideY, int M, int N, int K, int S, int heads) {
    __shared__ unsigned short Atile[16 * 32];

    int tid = threadIdx.x;
    int lane = tid & 31;
    int wave = tid >> 5;
    int m0 = blockIdx.x * 16;
    int n0 = (blockIdx.y * 8 + wave) * 16;
    int b = blockIdx.z;

    const float* Ab  = A  + (long)b * strideA;
    const float* A2b = A2 + (long)b * strideA2;

    v8f acc = zero8();
    for (int k0 = 0; k0 < K; k0 += 32) {
        const float* Ap;
        int ld, kk;
        if (k0 < ksplit) { Ap = Ab;  ld = lda;  kk = k0; }
        else             { Ap = A2b; ld = lda2; kk = k0 - ksplit; }

        // ---- stage A tile (16 rows x 32 k) into LDS as bf16, coalesced f4 ----
        if (tid < 128) {
            if (AMODE == 0) {
                int row = tid >> 3;           // 0..15  (m)
                int c4  = (tid & 7) << 2;     // 0..28  (k)
                f4 t = *(const f4*)(Ap + (long)(m0 + row) * ld + kk + c4);
#pragma unroll
                for (int j = 0; j < 4; ++j) Atile[row * 32 + c4 + j] = f2bf(t[j]);
            } else {
                int krow = tid >> 2;          // 0..31  (k)
                int m4   = (tid & 3) << 2;    // 0..12  (m)
                f4 t = *(const f4*)(Ap + (long)(kk + krow) * ld + m0 + m4);
#pragma unroll
                for (int j = 0; j < 4; ++j) Atile[(m4 + j) * 32 + krow] = f2bf(t[j]);
            }
        }
        // prefetch next bf16 weight chunk for this wave's n-row (speculative)
        if (k0 + 32 < K)
            __builtin_prefetch(W + (long)(n0 + (lane & 15)) * K + k0 + 32, 0, 0);
        __syncthreads();

        v16bf af = loadA_lds(Atile, lane);
        v16bf bf = loadB_Wh(W, K, n0, k0, lane);
        acc = WMMA_BF16(af, bf, acc);
        __syncthreads();
    }

    int col = lane & 15;
    int half = lane >> 4;

    if (EP == 3) {
        int t  = n0 >> 8;                 // 0=q 1=k 2=v (N=768, 256-aligned)
        int h  = (n0 >> 5) & (heads - 1);
        int d0 = n0 & 31;
        float* base = (t == 0) ? Y0 : (t == 1) ? Y1 : Y2;
        float* dst = base + ((long)b * heads + h) * (long)S * 32;
#pragma unroll
        for (int r = 0; r < 8; ++r) {
            int m = m0 + r + 8 * half;
            dst[(long)m * 32 + d0 + col] = acc[r];
        }
    } else {
        float bval = bias ? bias[n0 + col] : 0.f;
#pragma unroll
        for (int r = 0; r < 8; ++r) {
            int m = m0 + r + 8 * half;
            float v = acc[r] + bval;
            if (EP == 1) v = gelu_exact(v);
            if (EP == 2)
                Y0[(long)b * strideY + (long)(n0 + col) * M + m] = v;
            else
                Y0[(long)b * strideY + (long)m * N + (n0 + col)] = v;
        }
    }
}

// ---------------------------------------------------------------------------
// One 32-key attention tile (QK^T -> exp -> +P·V, +P·1 for the row sums).
//
// No running max, no cross-lane reductions: q/k are L2-normalized and q is
// pre-scaled by hd^-0.5, so every logit is a cosine similarity in
// [-0.1768, +0.1768]. exp cannot overflow and softmax is shift-invariant,
// so plain exp(s) + final sum-normalization is exactly the reference
// softmax. Row sums are computed on the matrix pipe as P x all-ones
// (C-fragment rows line up with the output fragment rows).
//
// MASKED only for the final partial tile (S % 32 != 0, global branch).
// ---------------------------------------------------------------------------
template <bool MASKED>
__device__ __forceinline__ void attn_tile(
    const v16bf& qa, const v16bf& onesb, const unsigned short* __restrict__ k,
    const unsigned short* __restrict__ v, int key0, int S, int lane,
    unsigned short* Plds, unsigned short* Vtile, int VLD,
    v8f& o0, v8f& o1, v8f& rsum) {
    int col = lane & 15;
    int half = lane >> 4;

    // ---- stage V tile: one bf16 row (64B) per lane ----
    {
        int key = key0 + lane;
        int kc = MASKED ? (key < (S - 1) ? key : (S - 1)) : key;
        const u16x8* p = (const u16x8*)(v + (long)kc * 32);
        u16x8 t0 = p[0], t1 = p[1], t2 = p[2], t3 = p[3];
        u16x8* dst = (u16x8*)(Vtile + lane * VLD);
        if (MASKED) {
            bool valid = key < S;
            dst[0] = sel8(valid, t0);
            dst[1] = sel8(valid, t1);
            dst[2] = sel8(valid, t2);
            dst[3] = sel8(valid, t3);
        } else {
            dst[0] = t0; dst[1] = t1; dst[2] = t2; dst[3] = t3;
        }
    }

    // ---- scores: QK^T, K-dim = hd = 32 in a single WMMA per 16 keys ----
    v16bf kb0 = loadB_keys<MASKED>(k, key0, S, lane);
    v16bf kb1 = loadB_keys<MASKED>(k, key0 + 16, S, lane);
    v8f s0 = WMMA_BF16(qa, kb0, zero8());
    v8f s1 = WMMA_BF16(qa, kb1, zero8());

    bool c0 = !MASKED || (key0 + col) < S;
    bool c1 = !MASKED || (key0 + 16 + col) < S;

#pragma unroll
    for (int r = 0; r < 8; ++r) {
        float p0 = __expf(s0[r]);
        float p1 = __expf(s1[r]);
        if (MASKED) {
            p0 = c0 ? p0 : 0.f;
            p1 = c1 ? p1 : 0.f;
        }
        int row = r + 8 * half;
        Plds[row * 32 + col]      = f2bf(p0);
        Plds[row * 32 + 16 + col] = f2bf(p1);
    }
    __syncthreads();
    v16bf pa  = loadA_lds(Plds, lane);              // P: 16 x 32 keys
    v16bf vb0 = loadB_lds(Vtile, VLD, 0, lane);     // V: 32 keys x 16 d
    v16bf vb1 = loadB_lds(Vtile, VLD, 16, lane);
    o0   = WMMA_BF16(pa, vb0, o0);
    o1   = WMMA_BF16(pa, vb1, o1);
    rsum = WMMA_BF16(pa, onesb, rsum);              // row sums on the matrix pipe
    __syncthreads();
}

// ---------------------------------------------------------------------------
// Attention, one 16-row query tile per wave (blockDim = 32).
// q pre-scaled by hd^-0.5, q/k L2-normalized; q,k,v pre-converted to bf16,
// layout [B, heads, S, 32].  Out: [B, S, C] f32 with c = h*32 + d.
// grid = (S/16, heads, B)
// ---------------------------------------------------------------------------
__global__ __launch_bounds__(32) void attn_bf16_wmma(
    const unsigned short* __restrict__ Q, const unsigned short* __restrict__ Km,
    const unsigned short* __restrict__ Vm, float* __restrict__ Out,
    int S, int heads, int C) {
    const int VLD = 40;  // padded V row stride (ushorts): 80B rows, 16B-aligned runs
    __shared__ unsigned short Plds[16 * 32];
    __shared__ unsigned short Vtile[32 * VLD];

    int lane = threadIdx.x;
    int qt = blockIdx.x, h = blockIdx.y, b = blockIdx.z;
    long bh = (long)b * heads + h;
    const unsigned short* q = Q  + bh * (long)S * 32;
    const unsigned short* k = Km + bh * (long)S * 32;
    const unsigned short* v = Vm + bh * (long)S * 32;

    int q0 = qt * 16;
    v16bf qa = loadA_bf16(q, 32, q0, lane);   // 16x32: rows=queries, K=head dim

    Frag16 ones;                              // all-ones B frag for row sums
#pragma unroll
    for (int i = 0; i < 16; ++i) ones.h[i] = 0x3F80;  // bf16(1.0)

    v8f o0 = zero8(), o1 = zero8(), rsum = zero8();

    // mask-free main loop over full 32-key tiles (all of the local branch)
    int nfull = S >> 5;
    for (int kt = 0; kt < nfull; ++kt)
        attn_tile<false>(qa, ones.v, k, v, kt * 32, S, lane, Plds, Vtile, VLD, o0, o1, rsum);
    // masked remainder tile (global branch: S = 144 -> 16 leftover keys)
    if (S & 31)
        attn_tile<true>(qa, ones.v, k, v, nfull * 32, S, lane, Plds, Vtile, VLD, o0, o1, rsum);

    int col = lane & 15;
    int half = lane >> 4;
    float* outb = Out + (long)b * S * C;
#pragma unroll
    for (int r = 0; r < 8; ++r) {
        float inv = 1.0f / rsum[r];
        int s = q0 + r + 8 * half;
        outb[(long)s * C + h * 32 + col]      = o0[r] * inv;
        outb[(long)s * C + h * 32 + 16 + col] = o1[r] * inv;
    }
}

// ---------------------------------------------------------------------------
// L2-normalize rows of 32, scale, emit bf16 (q: scale = hd^-0.5, k: 1)
// ---------------------------------------------------------------------------
__global__ void l2norm_rows32_bf16(const float* __restrict__ src,
                                   unsigned short* __restrict__ dst,
                                   long rows, float scale) {
    long i = (long)blockIdx.x * blockDim.x + threadIdx.x;
    if (i >= rows) return;
    const f4* p = (const f4*)(src + i * 32);
    f4 vals[8];
    float ss = 0.f;
#pragma unroll
    for (int g = 0; g < 8; ++g) {
        vals[g] = p[g];
#pragma unroll
        for (int j = 0; j < 4; ++j) ss += vals[g][j] * vals[g][j];
    }
    float inv = scale / fmaxf(sqrtf(ss), 1e-12f);
    u16x8* o = (u16x8*)(dst + i * 32);
#pragma unroll
    for (int g = 0; g < 4; ++g) {
        u16x8 t;
#pragma unroll
        for (int j = 0; j < 4; ++j) {
            t[j]     = f2bf(vals[2 * g][j] * inv);
            t[4 + j] = f2bf(vals[2 * g + 1][j] * inv);
        }
        o[g] = t;
    }
}

// elementwise f32 -> bf16 (n multiple of 8)
__global__ void f32_to_bf16(const float* __restrict__ src,
                            unsigned short* __restrict__ dst, long n) {
    long i = ((long)blockIdx.x * blockDim.x + threadIdx.x) * 8;
    if (i >= n) return;
    f4 a = *(const f4*)(src + i);
    f4 b = *(const f4*)(src + i + 4);
    u16x8 o;
#pragma unroll
    for (int j = 0; j < 4; ++j) { o[j] = f2bf(a[j]); o[4 + j] = f2bf(b[j]); }
    *(u16x8*)(dst + i) = o;
}

// 4x4 average pool: (B,C,48,48) -> (B,C,12,12)
__global__ void avgpool4(const float* __restrict__ X, float* __restrict__ Xc,
                         int B, int C, int H) {
    int Ho = H / 4;
    long total = (long)B * C * Ho * Ho;
    long idx = (long)blockIdx.x * blockDim.x + threadIdx.x;
    if (idx >= total) return;
    int xo = idx % Ho;
    long t = idx / Ho;
    int yo = t % Ho; t /= Ho;
    int c = t % C;
    int b = t / C;
    const float* src = X + (((long)b * C + c) * H + yo * 4) * H + xo * 4;
    float s = 0.f;
#pragma unroll
    for (int dy = 0; dy < 4; ++dy) {
        f4 row = *(const f4*)(src + dy * H);
        s += row[0] + row[1] + row[2] + row[3];
    }
    Xc[idx] = s * (1.0f / 16.0f);
}

// half-pixel bilinear: G (B, Hi*Hi, C) -> Out (B, Ho*Ho, C)
__global__ void upsample_bilinear(const float* __restrict__ G, float* __restrict__ Out,
                                  int B, int C, int Hi, int Ho) {
    long total = (long)B * Ho * Ho * C;
    long idx = (long)blockIdx.x * blockDim.x + threadIdx.x;
    if (idx >= total) return;
    int c = idx % C;
    long t = idx / C;
    int xo = t % Ho; t /= Ho;
    int yo = t % Ho;
    int b = t / Ho;
    float scale = (float)Hi / (float)Ho;
    float sy = (yo + 0.5f) * scale - 0.5f;
    float sx = (xo + 0.5f) * scale - 0.5f;
    int y0 = (int)floorf(sy), x0 = (int)floorf(sx);
    float wy = sy - y0, wx = sx - x0;
    int y0c = min(max(y0, 0), Hi - 1), y1c = min(max(y0 + 1, 0), Hi - 1);
    int x0c = min(max(x0, 0), Hi - 1), x1c = min(max(x0 + 1, 0), Hi - 1);
    const float* gb = G + (long)b * Hi * Hi * C;
    float v00 = gb[((long)y0c * Hi + x0c) * C + c];
    float v01 = gb[((long)y0c * Hi + x1c) * C + c];
    float v10 = gb[((long)y1c * Hi + x0c) * C + c];
    float v11 = gb[((long)y1c * Hi + x1c) * C + c];
    float v0 = v00 + wx * (v01 - v00);
    float v1 = v10 + wx * (v11 - v10);
    Out[idx] = v0 + wy * (v1 - v0);
}

// ---------------------------------------------------------------------------
// Launch: full forward pass
// ---------------------------------------------------------------------------
extern "C" void kernel_launch(void* const* d_in, const int* in_sizes, int n_in,
                              void* d_out, int out_size, void* d_ws, size_t ws_size,
                              hipStream_t stream) {
    (void)in_sizes; (void)n_in; (void)out_size; (void)ws_size;
    const float* x        = (const float*)d_in[0];
    const float* w_qkv_l  = (const float*)d_in[1];
    const float* w_proj_l = (const float*)d_in[2];
    const float* b_proj_l = (const float*)d_in[3];
    const float* w_qkv_g  = (const float*)d_in[4];
    const float* w_proj_g = (const float*)d_in[5];
    const float* b_proj_g = (const float*)d_in[6];
    const float* w_f1     = (const float*)d_in[7];
    const float* b_f1     = (const float*)d_in[8];
    const float* w_f2     = (const float*)d_in[9];
    const float* b_f2     = (const float*)d_in[10];
    float* out = (float*)d_out;

    const int B = 4, C = 256, H = 48, heads = 8;
    const int SL = H * H;             // 2304
    const int Hg = 12, SG = Hg * Hg;  // 144
    const long BIG = (long)B * SL * C;   // 2,359,296
    const long SML = (long)B * SG * C;   // 147,456
    const float qscale = 0.17677669529663688f;  // 32^-0.5

    float* ws = (float*)d_ws;
    float* Qb  = ws;              // f32 scratch
    float* Kb  = Qb + BIG;
    float* Vb  = Kb + BIG;
    float* AO  = Vb + BIG;
    float* LT  = AO + BIG;
    float* XC  = LT + BIG;
    float* QG  = XC + SML;
    float* KG  = QG + SML;
    float* VG  = KG + SML;
    float* AOG = VG + SML;
    float* GS  = AOG + SML;
    float* GT  = Qb;              // reuse: q/k/v f32 free after bf16 convert
    float* HB  = Kb;

    unsigned short* us = (unsigned short*)(ws + 5 * BIG + 6 * SML);
    unsigned short* QhL = us;            // bf16 q/k/v local
    unsigned short* KhL = QhL + BIG;
    unsigned short* VhL = KhL + BIG;
    unsigned short* QhG = VhL + BIG;     // bf16 q/k/v global
    unsigned short* KhG = QhG + SML;
    unsigned short* VhG = KhG + SML;
    unsigned short* WhQL = VhG + SML;             // w_qkv_l  768x256
    unsigned short* WhPL = WhQL + 768 * 256;      // w_proj_l 256x256
    unsigned short* WhQG = WhPL + 256 * 256;      // w_qkv_g  768x256
    unsigned short* WhPG = WhQG + 768 * 256;      // w_proj_g 256x256
    unsigned short* WhF1 = WhPG + 256 * 256;      // w_f1     256x512
    unsigned short* WhF2 = WhF1 + 256 * 512;      // w_f2     256x256

    auto cvt = [&](const float* s, unsigned short* d, long n) {
        f32_to_bf16<<<(int)((n / 8 + 255) / 256), 256, 0, stream>>>(s, d, n);
    };

    // 0) pre-convert all weights to bf16 (once)
    cvt(w_qkv_l, WhQL, 768 * 256);
    cvt(w_proj_l, WhPL, 256 * 256);
    cvt(w_qkv_g, WhQG, 768 * 256);
    cvt(w_proj_g, WhPG, 256 * 256);
    cvt(w_f1, WhF1, 256 * 512);
    cvt(w_f2, WhF2, 256 * 256);

    // 1) QKV local: x (C,S) x w_qkv_l -> q/k/v f32 [B,h,S,32]
    gemm_bf16_wmma<1, 3><<<dim3(SL / 16, 48 / 8, B), 256, 0, stream>>>(
        x, x, C, SL, SL, (long)C * SL, (long)C * SL, WhQL, nullptr,
        Qb, Kb, Vb, 0, SL, 3 * C, C, SL, heads);

    // 2) normalize q (scaled) and k -> bf16; convert v -> bf16
    {
        long rows = (long)B * heads * SL;
        int grid = (int)((rows + 255) / 256);
        l2norm_rows32_bf16<<<grid, 256, 0, stream>>>(Qb, QhL, rows, qscale);
        l2norm_rows32_bf16<<<grid, 256, 0, stream>>>(Kb, KhL, rows, 1.0f);
        cvt(Vb, VhL, BIG);
    }

    // 3) local attention -> AO (B,S,C)
    attn_bf16_wmma<<<dim3(SL / 16, heads, B), 32, 0, stream>>>(QhL, KhL, VhL, AO, SL, heads, C);

    // 4) local projection -> LT (B,S,C)
    gemm_bf16_wmma<0, 0><<<dim3(SL / 16, 16 / 8, B), 256, 0, stream>>>(
        AO, AO, C, C, C, (long)SL * C, (long)SL * C, WhPL, b_proj_l,
        LT, nullptr, nullptr, (long)SL * C, SL, C, C, SL, heads);

    // 5) avg pool -> XC (B,C,12,12)
    avgpool4<<<(int)((SML + 255) / 256), 256, 0, stream>>>(x, XC, B, C, H);

    // 6) QKV global
    gemm_bf16_wmma<1, 3><<<dim3(SG / 16, 48 / 8, B), 256, 0, stream>>>(
        XC, XC, C, SG, SG, (long)C * SG, (long)C * SG, WhQG, nullptr,
        QG, KG, VG, 0, SG, 3 * C, C, SG, heads);

    // 7) normalize / convert global q,k,v -> bf16
    {
        long rows = (long)B * heads * SG;
        int grid = (int)((rows + 255) / 256);
        l2norm_rows32_bf16<<<grid, 256, 0, stream>>>(QG, QhG, rows, qscale);
        l2norm_rows32_bf16<<<grid, 256, 0, stream>>>(KG, KhG, rows, 1.0f);
        cvt(VG, VhG, SML);
    }

    // 8) global attention -> AOG (B,144,C)
    attn_bf16_wmma<<<dim3(SG / 16, heads, B), 32, 0, stream>>>(QhG, KhG, VhG, AOG, SG, heads, C);

    // 9) global projection -> GS (B,144,C)
    gemm_bf16_wmma<0, 0><<<dim3(SG / 16, 16 / 8, B), 256, 0, stream>>>(
        AOG, AOG, C, C, C, (long)SG * C, (long)SG * C, WhPG, b_proj_g,
        GS, nullptr, nullptr, (long)SG * C, SG, C, C, SG, heads);

    // 10) bilinear upsample GS -> GT (B,S,C)
    upsample_bilinear<<<(int)((BIG + 255) / 256), 256, 0, stream>>>(GS, GT, B, C, Hg, H);

    // 11) fusion f1 over concat[LT | GT] (K=512), bias + exact GELU -> HB
    gemm_bf16_wmma<0, 1><<<dim3(SL / 16, 16 / 8, B), 256, 0, stream>>>(
        LT, GT, C, C, C, (long)SL * C, (long)SL * C, WhF1, b_f1,
        HB, nullptr, nullptr, (long)SL * C, SL, C, 2 * C, SL, heads);

    // 12) f2, bias, transposed store -> out (B,C,H,W)
    gemm_bf16_wmma<0, 2><<<dim3(SL / 16, 16 / 8, B), 256, 0, stream>>>(
        HB, HB, C, C, C, (long)SL * C, (long)SL * C, WhF2, b_f2,
        out, nullptr, nullptr, (long)C * SL, SL, C, C, SL, heads);
}
